// DeformBlock_76922864271948
// MI455X (gfx1250) — matrix operand order repaired and checked by reading, not compiled, not run on hardware
//
#include <hip/hip_runtime.h>
#include <hip/hip_bf16.h>
#include <stdint.h>
#include <stddef.h>

// ---------------------------------------------------------------------------
// Geometry (compile-time constants from the reference)
// ---------------------------------------------------------------------------
#define NB     4
#define CIN    1024
#define HH     64
#define WW     64
#define HWSZ   4096            // H*W
#define NTOT   16384           // B*H*W
#define PL     256             // PLANES
#define PL4    1024            // PLANES*4
#define KK9    9
#define KDEF   2304            // PL*9

typedef __attribute__((ext_vector_type(16))) __bf16 bf16x16;
typedef __attribute__((ext_vector_type(8)))  float  f32x8;

__device__ __forceinline__ int iclamp(int v, int lo, int hi) {
    return v < lo ? lo : (v > hi ? hi : v);
}

// ---------------------------------------------------------------------------
// CDNA5 async memory->LDS staging (no VGPR round trip, ASYNCcnt tracked).
// saddr form: mem = SGPR64 + VGPR_I32 + ioffset ; 16 bytes per lane.
// ---------------------------------------------------------------------------
__device__ __forceinline__ void async_ld_b128(uint32_t lds_off, uint32_t voff,
                                              const void* sbase) {
    asm volatile("global_load_async_to_lds_b128 %0, %1, %2"
                 :: "v"(lds_off), "v"(voff), "s"(sbase) : "memory");
}
__device__ __forceinline__ void wait_async0() { asm volatile("s_wait_asynccnt 0x0" ::: "memory"); }
__device__ __forceinline__ void wait_async4() { asm volatile("s_wait_asynccnt 0x4" ::: "memory"); }
__device__ __forceinline__ void wait_async5() { asm volatile("s_wait_asynccnt 0x5" ::: "memory"); }

__device__ __forceinline__ uint32_t lds_addr(const void* p) {
    return (uint32_t)(uintptr_t)p;          // low 32 bits of generic = LDS offset
}

// ---------------------------------------------------------------------------
// WMMA fragment helpers (wave32, v_wmma_f32_16x16x32_bf16).
// All LDS tiles use panel layout: element (row, k) at row*32 + k, k in [0,32).
// A 16x32 bf16: lanes 0-15 K base 0, lanes 16-31 K base 8.
// B 32x16 bf16 ([col][k]): lanes 0-15 K base 0, lanes 16-31 K base 8.
// ---------------------------------------------------------------------------
__device__ __forceinline__ bf16x16 frag_a_row(const __bf16* rowp, unsigned lane) {
    union { uint32_t u[8]; bf16x16 v; } r;
    const uint32_t* p = (const uint32_t*)(rowp + ((lane & 16u) ? 8 : 0));
#pragma unroll
    for (int i = 0; i < 4; ++i) { r.u[i] = p[i]; r.u[i + 4] = p[i + 8]; }
    return r.v;
}
__device__ __forceinline__ bf16x16 frag_b_col(const __bf16* colp, unsigned lane) {
    union { uint32_t u[8]; bf16x16 v; } r;
    const uint32_t* p = (const uint32_t*)(colp + ((lane & 16u) ? 8 : 0));
#pragma unroll
    for (int i = 0; i < 4; ++i) { r.u[i] = p[i]; r.u[i + 4] = p[i + 8]; }
    return r.v;
}
__device__ __forceinline__ f32x8 wmma_bf16(bf16x16 a, bf16x16 b, f32x8 c) {
    return __builtin_amdgcn_wmma_f32_16x16x32_bf16(false, a, false, b,
                                                   (short)0, c, false, false);
}

// ---------------------------------------------------------------------------
// Pack kernels.  Panel format for an MxK matrix: element (m,k) stored at
// (k>>5)*(M*32) + m*32 + (k&31)  -> every 32-wide K chunk is contiguous.
// ---------------------------------------------------------------------------

// x [B][CIN][HW] fp32 -> X panels [32][NTOT][32] bf16 (coalesced both sides)
__global__ void k_pack_x(const float* __restrict__ x, __bf16* __restrict__ Xp) {
    int i = blockIdx.x * 256 + threadIdx.x;           // 32 panels * NTOT
    if (i >= 32 * NTOT) return;
    int p = i >> 14, n = i & (NTOT - 1);
    int b = n >> 12, hw = n & (HWSZ - 1);
    const float* src = x + ((size_t)(b * CIN + p * 32) << 12) + hw;
    __bf16 tmp[32];
#pragma unroll
    for (int j = 0; j < 32; ++j) { tmp[j] = (__bf16)(*src); src += HWSZ; }
    uint4* dst = (uint4*)(Xp + (size_t)p * (NTOT * 32) + (size_t)n * 32);
    const uint4* s4 = (const uint4*)tmp;
#pragma unroll
    for (int j = 0; j < 4; ++j) dst[j] = s4[j];
}

// generic MxK fp32 row-major -> bf16 panels (K = 1<<Kshift)
__global__ void k_pack_wpanel(const float* __restrict__ w, __bf16* __restrict__ A,
                              int M, int Kshift) {
    int i = blockIdx.x * 256 + threadIdx.x;
    if (i >= (M << Kshift)) return;
    int m = i >> Kshift, c = i & ((1 << Kshift) - 1);
    A[(size_t)(c >> 5) * ((size_t)M * 32) + m * 32 + (c & 31)] = (__bf16)w[i];
}

// w2 [o][c][kh][kw] -> A2 panels of [o][kd], kd = k*256+c
__global__ void k_pack_w2(const float* __restrict__ w, __bf16* __restrict__ A) {
    int i = blockIdx.x * 256 + threadIdx.x;                 // PL*KDEF
    if (i >= PL * KDEF) return;
    int o = i / KDEF, r = i % KDEF;
    int k = r >> 8, c = r & 255;
    A[(size_t)(r >> 5) * (PL * 32) + o * 32 + (r & 31)] =
        (__bf16)w[(size_t)(o * PL + c) * KK9 + k];
}

// woff [18][c][kh][kw] -> Aoff [32][kd] row-major, rows 18..31 zero
__global__ void k_pack_woff(const float* __restrict__ w, __bf16* __restrict__ A) {
    int i = blockIdx.x * 256 + threadIdx.x;                 // 32*KDEF
    if (i >= 32 * KDEF) return;
    int o = i / KDEF, r = i % KDEF;
    int k = r >> 8, c = r & 255;
    A[i] = (o < 18) ? (__bf16)w[(size_t)(o * PL + c) * KK9 + k] : (__bf16)0.0f;
}

__global__ void k_pack_bn(const float* g1, const float* b1, const float* m1, const float* v1,
                          const float* g2, const float* b2, const float* m2, const float* v2,
                          const float* bconv2,
                          const float* g3, const float* b3, const float* m3, const float* v3,
                          float* s1, float* sh1, float* s2, float* sh2,
                          float* s3, float* sh3) {
    int i = blockIdx.x * 256 + threadIdx.x;
    if (i < 256) {
        float sc = g1[i] * rsqrtf(v1[i] + 1e-5f);
        s1[i] = sc; sh1[i] = b1[i] - m1[i] * sc;
    } else if (i < 512) {
        int j = i - 256;
        float sc = g2[j] * rsqrtf(v2[j] + 1e-5f);
        s2[j] = sc; sh2[j] = (bconv2[j] - m2[j]) * sc + b2[j];
    } else if (i < 1536) {
        int j = i - 512;
        float sc = g3[j] * rsqrtf(v3[j] + 1e-5f);
        s3[j] = sc; sh3[j] = b3[j] - m3[j] * sc;
    }
}

// ---------------------------------------------------------------------------
// GEMM 1: h1 = relu(bn1(W1 * X)), async double-buffered staging.
// Grid: 256 WGs (64-wide N tiles), 256 threads (8 waves).
// ---------------------------------------------------------------------------
__global__ __launch_bounds__(256) void k_gemm1(
        const __bf16* __restrict__ A, const __bf16* __restrict__ X,
        const float* __restrict__ scale, const float* __restrict__ shift,
        __bf16* __restrict__ H1) {
    __shared__ __bf16 lA[2][PL * 32];   // 2 x 16 KB
    __shared__ __bf16 lB[2][64 * 32];   // 2 x 4 KB
    const int t = threadIdx.x;
    const int n0 = blockIdx.x * 64;
    const int w = t >> 5, lane = t & 31;
    f32x8 acc[2][4] = {};

    auto issue = [&](int kc, int buf) {
        uint32_t la = lds_addr(&lA[buf][0]) + t * 64;
        uint32_t ga = (uint32_t)kc * 16384u + t * 64;         // bytes
#pragma unroll
        for (int i = 0; i < 4; ++i) async_ld_b128(la + i * 16, ga + i * 16, A);
        uint32_t lb = lds_addr(&lB[buf][0]) + t * 16;
        uint32_t gb = ((uint32_t)kc * NTOT + n0) * 64u + t * 16;
        async_ld_b128(lb, gb, X);
    };

    issue(0, 0);
    for (int kc = 0; kc < CIN / 32; ++kc) {
        const bool more = (kc + 1) < CIN / 32;
        if (more) issue(kc + 1, (kc + 1) & 1);
        if (more) wait_async5(); else wait_async0();
        __syncthreads();
        const __bf16* cA = lA[kc & 1];
        const __bf16* cB = lB[kc & 1];
#pragma unroll
        for (int mt = 0; mt < 2; ++mt) {
            bf16x16 va = frag_a_row(cA + (w * 32 + mt * 16 + (lane & 15)) * 32, lane);
#pragma unroll
            for (int nt = 0; nt < 4; ++nt) {
                bf16x16 vb = frag_b_col(cB + (nt * 16 + (lane & 15)) * 32, lane);
                acc[mt][nt] = wmma_bf16(va, vb, acc[mt][nt]);
            }
        }
        __syncthreads();
    }
    const int hi = (lane >> 4) & 1;
#pragma unroll
    for (int mt = 0; mt < 2; ++mt)
#pragma unroll
        for (int nt = 0; nt < 4; ++nt) {
            int n = n0 + nt * 16 + (lane & 15);
            __bf16 tmp[8];
#pragma unroll
            for (int i = 0; i < 8; ++i) {
                int p = w * 32 + mt * 16 + hi * 8 + i;
                float v = acc[mt][nt][i] * scale[p] + shift[p];
                tmp[i] = (__bf16)(v > 0.f ? v : 0.f);
            }
            *(uint4*)(H1 + (size_t)w * (NTOT * 32) + (size_t)n * 32 + mt * 16 + hi * 8) =
                *(const uint4*)tmp;
        }
}

// ---------------------------------------------------------------------------
// Offset conv (M padded 18->32), H1 read in panel format.
// Grid: 128 WGs (128-wide N tiles), 256 threads.
// ---------------------------------------------------------------------------
__global__ __launch_bounds__(256) void k_offconv(
        const __bf16* __restrict__ A,          // [32][2304] row-major
        const __bf16* __restrict__ H1,         // panels [8][NTOT][32]
        const float* __restrict__ boff,
        float* __restrict__ OFFS) {
    __shared__ __bf16 lA[32 * 32];      // 2 KB
    __shared__ __bf16 lB[128 * 32];     // 8 KB  ([col][k])
    const int t = threadIdx.x;
    const int n0 = blockIdx.x * 128;
    const int w = t >> 5, lane = t & 31;
    f32x8 acc[2] = {};

    for (int kc = 0; kc < KDEF / 32; ++kc) {
        const int tap = kc >> 3;
        if (t < 128) {
            int row = t >> 2, seg = (t & 3) * 8;
            *(uint4*)(lA + row * 32 + seg) =
                *(const uint4*)(A + (size_t)row * KDEF + kc * 32 + seg);
        }
        {
            int cn = t & 127, cg = t >> 7;          // 2 groups of 16 channels
            int n = n0 + cn;
            int b = n >> 12, hw = n & (HWSZ - 1);
            int hy = hw >> 6, wx = hw & 63;
            int yy = hy + (tap / 3) * 2 - 2;
            int xx = wx + (tap % 3) * 2 - 2;
            bool ok = (yy >= 0) & (yy < HH) & (xx >= 0) & (xx < WW);
            uint4 q0 = {0,0,0,0}, q1 = {0,0,0,0};
            if (ok) {
                size_t s = (size_t)((b << 12) + yy * 64 + xx);
                const __bf16* hp = H1 + (size_t)(kc & 7) * (NTOT * 32) + s * 32 + cg * 16;
                q0 = ((const uint4*)hp)[0];
                q1 = ((const uint4*)hp)[1];
            }
            uint4* d = (uint4*)(lB + cn * 32 + cg * 16);
            d[0] = q0; d[1] = q1;
        }
        __syncthreads();
        {
            bf16x16 vb = frag_b_col(lB + (w * 16 + (lane & 15)) * 32, lane);
#pragma unroll
            for (int mt = 0; mt < 2; ++mt) {
                bf16x16 va = frag_a_row(lA + (mt * 16 + (lane & 15)) * 32, lane);
                acc[mt] = wmma_bf16(va, vb, acc[mt]);
            }
        }
        __syncthreads();
    }
    int n = n0 + w * 16 + (lane & 15);
#pragma unroll
    for (int mt = 0; mt < 2; ++mt)
#pragma unroll
        for (int i = 0; i < 8; ++i) {
            int p = mt * 16 + ((lane >> 4) & 1) * 8 + i;
            if (p < 18) OFFS[(size_t)p * NTOT + n] = acc[mt][i] + boff[p];
        }
}

// ---------------------------------------------------------------------------
// Deformable conv GEMM: fused bilinear im2col in B staging; async A prefetch.
// Grid: 256 WGs, 256 threads.
// ---------------------------------------------------------------------------
__global__ __launch_bounds__(256) void k_deform(
        const __bf16* __restrict__ A,          // panels [72][PL][32]
        const __bf16* __restrict__ H1,         // panels [8][NTOT][32]
        const float* __restrict__ OFFS,        // [18][16384]
        const float* __restrict__ scale, const float* __restrict__ shift,
        __bf16* __restrict__ H2) {
    __shared__ __bf16 lA[2][PL * 32];
    __shared__ __bf16 lB[64 * 32];
    const int t = threadIdx.x;
    const int n0 = blockIdx.x * 64;
    const int w = t >> 5, lane = t & 31;
    f32x8 acc[2][4] = {};

    auto issueA = [&](int kc, int buf) {
        uint32_t la = lds_addr(&lA[buf][0]) + t * 64;
        uint32_t ga = (uint32_t)kc * 16384u + t * 64;
#pragma unroll
        for (int i = 0; i < 4; ++i) async_ld_b128(la + i * 16, ga + i * 16, A);
    };

    issueA(0, 0);
    for (int kc = 0; kc < KDEF / 32; ++kc) {
        const int tap = kc >> 3;
        { // B staging: bilinear sample 32 channels x 64 cols (panel gathers)
            const int cn = t & 63, cg = t >> 6;     // 4 groups of 8 channels
            const int n = n0 + cn;
            const int b = n >> 12, hw = n & (HWSZ - 1);
            const int hy = hw >> 6, wxp = hw & 63;
            const float offy = OFFS[(size_t)(tap * 2 + 0) * NTOT + n];
            const float offx = OFFS[(size_t)(tap * 2 + 1) * NTOT + n];
            const float py = (float)(hy + (tap / 3) * 2 - 2) + offy;
            const float px = (float)(wxp + (tap % 3) * 2 - 2) + offx;
            const float y0f = floorf(py), x0f = floorf(px);
            const float fy = py - y0f, fx = px - x0f;
            const int y0 = (int)y0f, x0 = (int)x0f;
            const int y1 = y0 + 1, x1 = x0 + 1;
            const bool vy0 = (y0 >= 0) & (y0 < HH), vy1 = (y1 >= 0) & (y1 < HH);
            const bool vx0 = (x0 >= 0) & (x0 < WW), vx1 = (x1 >= 0) & (x1 < WW);
            float w00 = (1.f - fy) * (1.f - fx), w01 = (1.f - fy) * fx;
            float w10 = fy * (1.f - fx),         w11 = fy * fx;
            if (!(vy0 & vx0)) w00 = 0.f;
            if (!(vy0 & vx1)) w01 = 0.f;
            if (!(vy1 & vx0)) w10 = 0.f;
            if (!(vy1 & vx1)) w11 = 0.f;
            const int y0c = iclamp(y0, 0, HH - 1), y1c = iclamp(y1, 0, HH - 1);
            const int x0c = iclamp(x0, 0, WW - 1), x1c = iclamp(x1, 0, WW - 1);
            const int base = b << 12;
            const __bf16* hp = H1 + (size_t)(kc & 7) * (NTOT * 32) + cg * 8;
            const uint4 q00 = *(const uint4*)(hp + (size_t)(base + y0c * 64 + x0c) * 32);
            const uint4 q01 = *(const uint4*)(hp + (size_t)(base + y0c * 64 + x1c) * 32);
            const uint4 q10 = *(const uint4*)(hp + (size_t)(base + y1c * 64 + x0c) * 32);
            const uint4 q11 = *(const uint4*)(hp + (size_t)(base + y1c * 64 + x1c) * 32);
            const __bf16* e00 = (const __bf16*)&q00;
            const __bf16* e01 = (const __bf16*)&q01;
            const __bf16* e10 = (const __bf16*)&q10;
            const __bf16* e11 = (const __bf16*)&q11;
            __bf16 tmp[8];
#pragma unroll
            for (int j = 0; j < 8; ++j) {
                float v = w00 * (float)e00[j] + w01 * (float)e01[j]
                        + w10 * (float)e10[j] + w11 * (float)e11[j];
                tmp[j] = (__bf16)v;
            }
            *(uint4*)(lB + cn * 32 + cg * 8) = *(const uint4*)tmp;
        }
        const bool more = (kc + 1) < KDEF / 32;
        if (more) issueA(kc + 1, (kc + 1) & 1);
        if (more) wait_async4(); else wait_async0();
        __syncthreads();
        const __bf16* cA = lA[kc & 1];
#pragma unroll
        for (int mt = 0; mt < 2; ++mt) {
            bf16x16 va = frag_a_row(cA + (w * 32 + mt * 16 + (lane & 15)) * 32, lane);
#pragma unroll
            for (int nt = 0; nt < 4; ++nt) {
                bf16x16 vb = frag_b_col(lB + (nt * 16 + (lane & 15)) * 32, lane);
                acc[mt][nt] = wmma_bf16(va, vb, acc[mt][nt]);
            }
        }
        __syncthreads();
    }
    const int hi = (lane >> 4) & 1;
#pragma unroll
    for (int mt = 0; mt < 2; ++mt)
#pragma unroll
        for (int nt = 0; nt < 4; ++nt) {
            int n = n0 + nt * 16 + (lane & 15);
            __bf16 tmp[8];
#pragma unroll
            for (int i = 0; i < 8; ++i) {
                int p = w * 32 + mt * 16 + hi * 8 + i;
                float v = acc[mt][nt][i] * scale[p] + shift[p];
                tmp[i] = (__bf16)(v > 0.f ? v : 0.f);
            }
            *(uint4*)(H2 + (size_t)w * (NTOT * 32) + (size_t)n * 32 + mt * 16 + hi * 8) =
                *(const uint4*)tmp;
        }
}

// ---------------------------------------------------------------------------
// GEMM 2: out = relu(bn3(W3 * h2) + x), async double-buffered staging.
// Grid: (256 N tiles) x (4 M groups of 256 rows), 256 threads.
// ---------------------------------------------------------------------------
__global__ __launch_bounds__(256) void k_gemm2(
        const __bf16* __restrict__ A,          // panels [8][PL4][32]
        const __bf16* __restrict__ H2,         // panels [8][NTOT][32]
        const float* __restrict__ xres,
        const float* __restrict__ scale, const float* __restrict__ shift,
        float* __restrict__ OUT) {
    __shared__ __bf16 lA[2][PL * 32];
    __shared__ __bf16 lB[2][64 * 32];
    const int t = threadIdx.x;
    const int n0 = blockIdx.x * 64;
    const int mg = blockIdx.y;                 // 0..3
    const int w = t >> 5, lane = t & 31;
    f32x8 acc[2][4] = {};

    auto issue = [&](int kc, int buf) {
        uint32_t la = lds_addr(&lA[buf][0]) + t * 64;
        uint32_t ga = (uint32_t)kc * 65536u + (uint32_t)mg * 16384u + t * 64;
#pragma unroll
        for (int i = 0; i < 4; ++i) async_ld_b128(la + i * 16, ga + i * 16, A);
        uint32_t lb = lds_addr(&lB[buf][0]) + t * 16;
        uint32_t gb = ((uint32_t)kc * NTOT + n0) * 64u + t * 16;
        async_ld_b128(lb, gb, H2);
    };

    issue(0, 0);
    for (int kc = 0; kc < PL / 32; ++kc) {
        const bool more = (kc + 1) < PL / 32;
        if (more) issue(kc + 1, (kc + 1) & 1);
        if (more) wait_async5(); else wait_async0();
        __syncthreads();
        const __bf16* cA = lA[kc & 1];
        const __bf16* cB = lB[kc & 1];
#pragma unroll
        for (int mt = 0; mt < 2; ++mt) {
            bf16x16 va = frag_a_row(cA + (w * 32 + mt * 16 + (lane & 15)) * 32, lane);
#pragma unroll
            for (int nt = 0; nt < 4; ++nt) {
                bf16x16 vb = frag_b_col(cB + (nt * 16 + (lane & 15)) * 32, lane);
                acc[mt][nt] = wmma_bf16(va, vb, acc[mt][nt]);
            }
        }
        __syncthreads();
    }
    const int hi = (lane >> 4) & 1;
#pragma unroll
    for (int mt = 0; mt < 2; ++mt)
#pragma unroll
        for (int nt = 0; nt < 4; ++nt) {
            int n = n0 + nt * 16 + (lane & 15);
            int b = n >> 12, hw = n & (HWSZ - 1);
#pragma unroll
            for (int i = 0; i < 8; ++i) {
                int p = mg * 256 + w * 32 + mt * 16 + hi * 8 + i;
                size_t oi = ((size_t)(b * PL4 + p) << 12) + hw;
                float v = acc[mt][nt][i] * scale[p] + shift[p] + xres[oi];
                OUT[oi] = v > 0.f ? v : 0.f;
            }
        }
}

// ---------------------------------------------------------------------------
// Host-side launch
// ---------------------------------------------------------------------------
extern "C" void kernel_launch(void* const* d_in, const int* in_sizes, int n_in,
                              void* d_out, int out_size, void* d_ws, size_t ws_size,
                              hipStream_t stream) {
    const float* x      = (const float*)d_in[0];
    const float* w1     = (const float*)d_in[1];
    const float* gamma1 = (const float*)d_in[2];
    const float* beta1  = (const float*)d_in[3];
    const float* mean1  = (const float*)d_in[4];
    const float* var1   = (const float*)d_in[5];
    const float* woff   = (const float*)d_in[6];
    const float* boff   = (const float*)d_in[7];
    const float* w2     = (const float*)d_in[8];
    const float* bconv2 = (const float*)d_in[9];
    const float* gamma2 = (const float*)d_in[10];
    const float* beta2  = (const float*)d_in[11];
    const float* mean2  = (const float*)d_in[12];
    const float* var2   = (const float*)d_in[13];
    const float* w3     = (const float*)d_in[14];
    const float* gamma3 = (const float*)d_in[15];
    const float* beta3  = (const float*)d_in[16];
    const float* mean3  = (const float*)d_in[17];
    const float* var3   = (const float*)d_in[18];
    float* out = (float*)d_out;

    char* ws = (char*)d_ws;
    __bf16* X    = (__bf16*)(ws + 0);                 // panels, 32 MB
    __bf16* H1   = (__bf16*)(ws + 33554432);          // panels, 8 MB
    __bf16* H2   = (__bf16*)(ws + 41943040);          // panels, 8 MB
    float*  OFFS = (float*) (ws + 50331648);          // 18*16384*4
    __bf16* A1   = (__bf16*)(ws + 51511296);          // panels, 512 KB
    __bf16* A2   = (__bf16*)(ws + 52035584);          // panels, 1.125 MB
    __bf16* AOF  = (__bf16*)(ws + 53215232);          // row-major, 144 KB
    __bf16* A3   = (__bf16*)(ws + 53362688);          // panels, 512 KB
    float*  s1   = (float*) (ws + 53886976);
    float*  sh1  = s1 + 256;
    float*  s2   = sh1 + 256;
    float*  sh2  = s2 + 256;
    float*  s3   = sh2 + 256;
    float*  sh3  = s3 + 1024;

    // pack
    k_pack_bn<<<6, 256, 0, stream>>>(gamma1, beta1, mean1, var1,
                                     gamma2, beta2, mean2, var2, bconv2,
                                     gamma3, beta3, mean3, var3,
                                     s1, sh1, s2, sh2, s3, sh3);
    k_pack_wpanel<<<(PL * CIN + 255) / 256, 256, 0, stream>>>(w1, A1, PL, 10);
    k_pack_w2<<<(PL * KDEF + 255) / 256, 256, 0, stream>>>(w2, A2);
    k_pack_woff<<<(32 * KDEF + 255) / 256, 256, 0, stream>>>(woff, AOF);
    k_pack_wpanel<<<(PL4 * PL + 255) / 256, 256, 0, stream>>>(w3, A3, PL4, 8);
    k_pack_x<<<(32 * NTOT + 255) / 256, 256, 0, stream>>>(x, X);

    // fused pipeline
    k_gemm1<<<NTOT / 64, 256, 0, stream>>>(A1, X, s1, sh1, H1);
    k_offconv<<<NTOT / 128, 256, 0, stream>>>(AOF, H1, boff, OFFS);
    k_deform<<<NTOT / 64, 256, 0, stream>>>(A2, H1, OFFS, s2, sh2, H2);
    k_gemm2<<<dim3(NTOT / 64, 4), 256, 0, stream>>>(A3, H2, x, s3, sh3, out);

    (void)in_sizes; (void)n_in; (void)out_size; (void)ws_size;
}
